// MultiHeadAttentionNew_19679540150929
// MI455X (gfx1250) — compile-verified, matrix operand order checked by reading, not run on hardware
//
#include <hip/hip_runtime.h>
#include <hip/hip_bf16.h>
#include <math.h>

// Problem constants (from reference)
#define HN 4
#define BN 2
#define GN 2048
#define DN 128
#define KDN 32
#define EN 128

typedef __attribute__((ext_vector_type(16))) _Float16 v16h;
typedef __attribute__((ext_vector_type(8)))  float    v8f;
typedef __attribute__((ext_vector_type(4)))  float    f32x4;

// ---------------------------------------------------------------------------
// Kernel 1: Q/K/V projections with v_wmma_f32_16x16x32_f16.
//   Q,K stored f16 row-major [h][b][row][kd]   (A/B-operand friendly)
//   V  stored f16 TRANSPOSED [h][b][v][p]      (A-operand of the AV WMMA)
// One wave per (h,b,q-tile); 24 WMMAs per wave.
// ---------------------------------------------------------------------------
__global__ __launch_bounds__(32) void qkv_proj_kernel(
    const float* __restrict__ hsrc,
    const float* __restrict__ Wq, const float* __restrict__ Wk,
    const float* __restrict__ Wv,
    _Float16* __restrict__ Qh, _Float16* __restrict__ Kh,
    _Float16* __restrict__ Vt)
{
    const int id   = blockIdx.x;        // H*B*(G/16)
    const int qt   = id & 127;
    const int b    = (id >> 7) & (BN - 1);
    const int hh   = id >> 8;
    const int lane = threadIdx.x;
    const int lo   = lane & 15, hi = lane >> 4;
    const int q0   = qt * 16;

    // A tiles: 16 rows of h in f16 A-layout, 4 K-steps of 32
    v16h A[4];
    const float* hrow = hsrc + ((size_t)b * GN + q0 + lo) * DN;
    #pragma unroll
    for (int k = 0; k < 4; ++k) {
        const int d0 = k * 32 + hi * 8;
        #pragma unroll
        for (int i = 0; i < 8; ++i) {
            A[k][i]     = (_Float16)hrow[d0 + i];
            A[k][8 + i] = (_Float16)hrow[d0 + 16 + i];
        }
    }

    const float* Ws[3] = { Wq + hh * DN * KDN, Wk + hh * DN * KDN, Wv + hh * DN * KDN };
    #pragma unroll
    for (int m = 0; m < 3; ++m) {
        #pragma unroll
        for (int nt = 0; nt < 2; ++nt) {
            const int col = nt * 16 + lo;
            v8f c = {};
            #pragma unroll
            for (int k = 0; k < 4; ++k) {
                const int d0 = k * 32 + hi * 8;
                v16h Bt;
                #pragma unroll
                for (int i = 0; i < 8; ++i) {
                    Bt[i]     = (_Float16)Ws[m][(d0 + i) * KDN + col];
                    Bt[8 + i] = (_Float16)Ws[m][(d0 + 16 + i) * KDN + col];
                }
                c = __builtin_amdgcn_wmma_f32_16x16x32_f16(
                        false, A[k], false, Bt, (short)0, c, false, false);
            }
            if (m < 2) {  // Q, K: row-major f16
                _Float16* dst = (m == 0 ? Qh : Kh) + ((size_t)hh * BN + b) * GN * KDN;
                #pragma unroll
                for (int r = 0; r < 8; ++r)
                    dst[(size_t)(q0 + r + 8 * hi) * KDN + col] = (_Float16)c[r];
            } else {      // V: transposed f16 (v-major, p contiguous)
                _Float16* dst = Vt + (((size_t)hh * BN + b) * KDN + col) * GN;
                #pragma unroll
                for (int r = 0; r < 8; ++r)
                    dst[q0 + r + 8 * hi] = (_Float16)c[r];
            }
        }
    }
}

// ---------------------------------------------------------------------------
// Kernel 2: fused  scores -> concat(OSA) -> MLP(8->8->4) -> online softmax
//           -> attn*V (WMMA) -> Wout, plus the OSA pass-through copy.
// Grid: B*(G/16) blocks of 128 threads (4 waves). Wave w streams
// p in [w*512, w*512+512) in 32-wide chunks.
//
// OSA staging uses the CDNA5 async global->LDS path (ASYNCcnt), double
// buffered: chunk n+1 streams into buffer B while the wave computes on
// buffer A; a single s_wait_asynccnt 0x10 gates consumption. The 268 MB
// pass-through copy is re-read from LDS (free bandwidth) and written with
// non-temporal stores so the stream does not thrash L2-resident K/Vt/Wout.
// ---------------------------------------------------------------------------
#define WAVES  4
#define CHUNKS 16            // 2048 / (WAVES*32)
#define WLDS   (4 * 16 * 36) // per-wave staging floats (4ch x 16 x 32, stride 36)

__global__ __launch_bounds__(128) void fused_attn_kernel(
    const float* __restrict__ osa,
    const _Float16* __restrict__ Qh, const _Float16* __restrict__ Kh,
    const _Float16* __restrict__ Vt,
    const float* __restrict__ Wout,
    const float* __restrict__ w1, const float* __restrict__ b1,
    const float* __restrict__ w2, const float* __restrict__ b2,
    float* __restrict__ out, float* __restrict__ osa_copy)
{
    extern __shared__ float smem[];  // [2][WAVES][WLDS] + M/L stats
    const int tid  = threadIdx.x;
    const int lane = tid & 31;
    const int wave = tid >> 5;
    const int lo   = lane & 15, hi = lane >> 4;
    const int qt   = blockIdx.x & 127;
    const int b    = blockIdx.x >> 7;
    const int q0   = qt * 16;

    float* bufA = smem + wave * WLDS;                 // double buffers (wave-private)
    float* bufB = smem + WAVES * WLDS + wave * WLDS;
    float* ldsM = smem + 2 * WAVES * WLDS;            // [w][c][q]
    float* ldsL = ldsM + WAVES * 4 * 16;

    // MLP parameters (uniform -> scalar loads)
    float w1r[8][8], w2r[4][8], b1r[8], b2r[4];
    #pragma unroll
    for (int j = 0; j < 8; ++j) {
        b1r[j] = b1[j];
        #pragma unroll
        for (int c = 0; c < 8; ++c) w1r[j][c] = w1[j * 8 + c];
    }
    #pragma unroll
    for (int k = 0; k < 4; ++k) {
        b2r[k] = b2[k];
        #pragma unroll
        for (int j = 0; j < 8; ++j) w2r[k][j] = w2[k * 8 + j];
    }

    // Q tiles as WMMA B operand (col = q), one per head
    v16h Qb[HN];
    #pragma unroll
    for (int hd = 0; hd < HN; ++hd) {
        const _Float16* qrow = Qh + (((size_t)hd * BN + b) * GN + q0 + lo) * KDN;
        ((f32x4*)&Qb[hd])[0] = *(const f32x4*)(qrow + 8 * hi);
        ((f32x4*)&Qb[hd])[1] = *(const f32x4*)(qrow + 16 + 8 * hi);
    }

    v8f   acc[4][2];
    float mrun[4], lrun[4];
    #pragma unroll
    for (int k = 0; k < 4; ++k) {
        acc[k][0] = (v8f){}; acc[k][1] = (v8f){};
        mrun[k] = -3.0e38f; lrun[k] = 0.0f;
    }

    // Issue one chunk (16 async b128 ops) of OSA into an LDS buffer.
    // Per-lane LDS dest address = low 32 bits of the flat shared pointer.
    auto issue_async = [&](int chnk, float* buf) {
        #pragma unroll
        for (int c = 0; c < 4; ++c) {
            const size_t base = (((size_t)c * BN + b) * GN + q0) * GN
                              + (size_t)wave * 512 + (size_t)chnk * 32;
            const float* src = osa + base;
            float*       ld  = buf + c * (16 * 36);
            #pragma unroll
            for (int it = 0; it < 4; ++it) {
                const int row = it * 4 + (lane >> 3);
                const int col = (lane & 7) * 4;
                const unsigned la = (unsigned)(uintptr_t)(ld + row * 36 + col);
                const float*   ga = src + (size_t)row * GN + col;
                asm volatile("global_load_async_to_lds_b128 %0, %1, off"
                             :: "v"(la), "v"(ga) : "memory");
            }
        }
    };

    issue_async(0, bufA);

    for (int ch = 0; ch < CHUNKS; ++ch) {
        const int    pbase = wave * 512 + ch * 32;
        float* const cur   = (ch & 1) ? bufB : bufA;

        // Prefetch next chunk into the other buffer, then gate on this one.
        if (ch + 1 < CHUNKS) {
            issue_async(ch + 1, (ch & 1) ? bufA : bufB);
            asm volatile("s_wait_asynccnt 0x10" ::: "memory");
        } else {
            asm volatile("s_wait_asynccnt 0x0" ::: "memory");
        }

        // ---- Pass-through copy: LDS -> non-temporal global store
        #pragma unroll
        for (int c = 0; c < 4; ++c) {
            const size_t base = (((size_t)c * BN + b) * GN + q0) * GN + pbase;
            float* dst = osa_copy + base;
            float* ld  = cur + c * (16 * 36);
            #pragma unroll
            for (int it = 0; it < 4; ++it) {
                const int row = it * 4 + (lane >> 3);
                const int col = (lane & 7) * 4;
                f32x4 v = *(const f32x4*)(ld + row * 36 + col);
                __builtin_nontemporal_store(v, (f32x4*)(dst + (size_t)row * GN + col));
            }
        }

        // ---- Transposed scores: sT = K_tile x Q  (C layout: M=p, N=q)
        v8f sT[2][4];
        #pragma unroll
        for (int st = 0; st < 2; ++st)
            #pragma unroll
            for (int hd = 0; hd < HN; ++hd) {
                const _Float16* krow =
                    Kh + (((size_t)hd * BN + b) * GN + pbase + st * 16 + lo) * KDN;
                v16h Ka;
                ((f32x4*)&Ka)[0] = *(const f32x4*)(krow + 8 * hi);
                ((f32x4*)&Ka)[1] = *(const f32x4*)(krow + 16 + 8 * hi);
                v8f z = {};
                sT[st][hd] = __builtin_amdgcn_wmma_f32_16x16x32_f16(
                                 false, Ka, false, Qb[hd], (short)0, z, false, false);
            }

        // ---- Per-element MLP (channels: 4 scores + 4 OSA) -> y[4]
        float ef[4][16];
        float tmax[4] = { -3.0e38f, -3.0e38f, -3.0e38f, -3.0e38f };
        #pragma unroll
        for (int st = 0; st < 2; ++st)
            #pragma unroll
            for (int r = 0; r < 8; ++r) {
                const int pl = st * 16 + 8 * hi + r;
                float x[8];
                #pragma unroll
                for (int hd = 0; hd < 4; ++hd) x[hd] = sT[st][hd][r];
                #pragma unroll
                for (int c = 0; c < 4; ++c)
                    x[4 + c] = cur[(c * 16 + lo) * 36 + pl];
                float hid[8];
                #pragma unroll
                for (int j = 0; j < 8; ++j) {
                    float a = b1r[j];
                    #pragma unroll
                    for (int c = 0; c < 8; ++c) a = fmaf(w1r[j][c], x[c], a);
                    hid[j] = fmaxf(a, 0.0f);
                }
                #pragma unroll
                for (int k = 0; k < 4; ++k) {
                    float y = b2r[k];
                    #pragma unroll
                    for (int j = 0; j < 8; ++j) y = fmaf(w2r[k][j], hid[j], y);
                    ef[k][st * 8 + r] = y;
                    tmax[k] = fmaxf(tmax[k], y);
                }
            }

        // ---- Online softmax update; exp'ed tile repacks directly into the
        //      f16 B-operand layout of the AV WMMA (no cross-lane movement).
        v16h eB[4];
        #pragma unroll
        for (int k = 0; k < 4; ++k) {
            const float om    = __shfl_xor(tmax[k], 16);
            const float mnew  = fmaxf(mrun[k], fmaxf(tmax[k], om));
            const float scale = __expf(mrun[k] - mnew);
            mrun[k] = mnew;
            float s = 0.0f;
            #pragma unroll
            for (int i = 0; i < 16; ++i) {
                const float e = __expf(ef[k][i] - mnew);
                ef[k][i] = e;
                s += e;
            }
            lrun[k] = lrun[k] * scale + s + __shfl_xor(s, 16);
            acc[k][0] = acc[k][0] * scale;
            acc[k][1] = acc[k][1] * scale;
            #pragma unroll
            for (int i = 0; i < 16; ++i) eB[k][i] = (_Float16)ef[k][i];
        }

        // ---- AV: acc += Vt_tile (A, M=v K=p) x e (B, K=p N=q)
        #pragma unroll
        for (int k = 0; k < 4; ++k)
            #pragma unroll
            for (int vt = 0; vt < 2; ++vt) {
                const _Float16* vrow =
                    Vt + (((size_t)k * BN + b) * KDN + vt * 16 + lo) * GN + pbase;
                v16h Va;
                ((f32x4*)&Va)[0] = *(const f32x4*)(vrow + 8 * hi);
                ((f32x4*)&Va)[1] = *(const f32x4*)(vrow + 16 + 8 * hi);
                acc[k][vt] = __builtin_amdgcn_wmma_f32_16x16x32_f16(
                                 false, Va, false, eB[k], (short)0, acc[k][vt], false, false);
            }
    }

    // ---- Dump per-wave partials (reuse staging buffer A; asyncs drained) ----
    #pragma unroll
    for (int k = 0; k < 4; ++k) {
        #pragma unroll
        for (int vt = 0; vt < 2; ++vt)
            #pragma unroll
            for (int r = 0; r < 8; ++r)
                bufA[(k * 16 + lo) * 36 + vt * 16 + 8 * hi + r] = acc[k][vt][r];
        if (lane < 16) {
            ldsM[(wave * 4 + k) * 16 + lo] = mrun[k];
            ldsL[(wave * 4 + k) * 16 + lo] = lrun[k];
        }
    }
    __syncthreads();

    // ---- Combine waves + softmax normalize + Wout projection ---------------
    {
        const int q  = tid >> 3;
        const int e0 = (tid & 7) * 16;
        float o[16];
        #pragma unroll
        for (int e = 0; e < 16; ++e) o[e] = 0.0f;

        #pragma unroll
        for (int k = 0; k < 4; ++k) {
            float M = -3.0e38f;
            #pragma unroll
            for (int w = 0; w < WAVES; ++w)
                M = fmaxf(M, ldsM[(w * 4 + k) * 16 + q]);
            float L = 0.0f, ew[WAVES];
            #pragma unroll
            for (int w = 0; w < WAVES; ++w) {
                ew[w] = __expf(ldsM[(w * 4 + k) * 16 + q] - M);
                L += ldsL[(w * 4 + k) * 16 + q] * ew[w];
            }
            const float invL = 1.0f / L;
            #pragma unroll
            for (int v = 0; v < 32; ++v) {
                float hval = 0.0f;
                #pragma unroll
                for (int w = 0; w < WAVES; ++w)
                    hval += smem[w * WLDS + (k * 16 + q) * 36 + v] * ew[w];
                hval *= invL;
                const float* wo = Wout + ((size_t)k * KDN + v) * EN + e0;
                #pragma unroll
                for (int e = 0; e < 16; e += 4) {
                    f32x4 wv = *(const f32x4*)(wo + e);
                    o[e + 0] = fmaf(hval, wv[0], o[e + 0]);
                    o[e + 1] = fmaf(hval, wv[1], o[e + 1]);
                    o[e + 2] = fmaf(hval, wv[2], o[e + 2]);
                    o[e + 3] = fmaf(hval, wv[3], o[e + 3]);
                }
            }
        }
        float* op = out + ((size_t)b * GN + q0 + q) * EN + e0;
        #pragma unroll
        for (int e = 0; e < 16; e += 4)
            *(f32x4*)(op + e) = (f32x4){ o[e], o[e + 1], o[e + 2], o[e + 3] };
    }
}

// ---------------------------------------------------------------------------
extern "C" void kernel_launch(void* const* d_in, const int* in_sizes, int n_in,
                              void* d_out, int out_size, void* d_ws, size_t ws_size,
                              hipStream_t stream)
{
    const float* h    = (const float*)d_in[0];
    const float* osa  = (const float*)d_in[1];
    const float* Wq   = (const float*)d_in[2];
    const float* Wk   = (const float*)d_in[3];
    const float* Wv   = (const float*)d_in[4];
    const float* Wout = (const float*)d_in[5];
    const float* w1   = (const float*)d_in[6];
    const float* b1   = (const float*)d_in[7];
    const float* w2   = (const float*)d_in[8];
    const float* b2   = (const float*)d_in[9];

    float* out      = (float*)d_out;
    float* osa_copy = out + (size_t)BN * GN * EN;   // tuple element 2, flat

    // Workspace: f16 Q, K, V^T  (3 * 1 MB)
    _Float16* Qh = (_Float16*)d_ws;
    _Float16* Kh = Qh + (size_t)HN * BN * GN * KDN;
    _Float16* Vt = Kh + (size_t)HN * BN * GN * KDN;

    qkv_proj_kernel<<<HN * BN * (GN / 16), 32, 0, stream>>>(h, Wq, Wk, Wv, Qh, Kh, Vt);

    const size_t smemBytes = (2 * WAVES * WLDS + WAVES * 4 * 16 * 2) * sizeof(float);
    fused_attn_kernel<<<BN * (GN / 16), 128, smemBytes, stream>>>(
        osa, Qh, Kh, Vt, Wout, w1, b1, w2, b2, out, osa_copy);
}
